// RankAttentionDeblureNet_83915071029455
// MI455X (gfx1250) — matrix operand order repaired and checked by reading, not compiled, not run on hardware
//
#include <hip/hip_runtime.h>

typedef unsigned short u16;
typedef __attribute__((ext_vector_type(16))) __bf16 bf16x16;
typedef __attribute__((ext_vector_type(8)))  float  floatx8;

#define BATCH 4
#define CCH   256
#define HH    64
#define WW    64
#define LL    4096          // H*W
#define KPAD  264           // 256 + 8, keeps 16B alignment, staggers banks

__device__ __forceinline__ u16 f2bf(float f) {
    unsigned int u = __float_as_uint(f);
    unsigned int r = u + 0x7FFFu + ((u >> 16) & 1u);   // round to nearest even
    return (u16)(r >> 16);
}

// ---------------------------------------------------------------------------
// Kernel 1: transpose [C][L] -> [L][C] in bf16 for blur & current, plus
// per-pixel feature energy E[p] = sum_c blur[c,p]^2.
// grid: BATCH * (LL/64) blocks, 256 threads.
// ---------------------------------------------------------------------------
__global__ __launch_bounds__(256) void pack_kernel(
    const float* __restrict__ xblur, const float* __restrict__ xcur,
    u16* __restrict__ Xb, u16* __restrict__ Xc, float* __restrict__ E)
{
    const int b  = blockIdx.x >> 6;
    const int p0 = (blockIdx.x & 63) << 6;
    const int t  = threadIdx.x;
    const int tp = t & 63;       // pixel within tile
    const int tc = t >> 6;       // 0..3 channel phase

    __shared__ u16  sb[64 * KPAD];
    __shared__ u16  sc[64 * KPAD];
    __shared__ float pe[4][64];

    float eb = 0.f;
#pragma unroll 4
    for (int i = 0; i < 64; ++i) {
        int c = i * 4 + tc;
        size_t gi = ((size_t)(b * CCH + c)) * LL + p0 + tp;
        float vb = xblur[gi];
        float vc = xcur[gi];
        eb += vb * vb;
        sb[tp * KPAD + c] = f2bf(vb);
        sc[tp * KPAD + c] = f2bf(vc);
    }
    pe[tc][tp] = eb;
    __syncthreads();

    // coalesced write-out: thread -> (pixel = t>>2, 64-ch chunk = (t&3)*64)
    const int p  = t >> 2;
    const int cb = (t & 3) * 64;
    uint4* db = (uint4*)(Xb + ((size_t)(b * LL + p0 + p)) * CCH + cb);
    uint4* dc = (uint4*)(Xc + ((size_t)(b * LL + p0 + p)) * CCH + cb);
    const uint4* srb = (const uint4*)&sb[p * KPAD + cb];
    const uint4* src = (const uint4*)&sc[p * KPAD + cb];
#pragma unroll
    for (int i = 0; i < 8; ++i) { db[i] = srb[i]; dc[i] = src[i]; }

    if (t < 64)
        E[b * LL + p0 + t] = pe[0][t] + pe[1][t] + pe[2][t] + pe[3][t];
}

// ---------------------------------------------------------------------------
// Kernel 2: invn[b,l] = 1 / max(sqrt(sum of E over valid 3x3 nbhd), 1e-12)
// ---------------------------------------------------------------------------
__global__ __launch_bounds__(256) void invn_kernel(
    const float* __restrict__ E, float* __restrict__ invn)
{
    int g = blockIdx.x * 256 + threadIdx.x;
    if (g >= BATCH * LL) return;
    int b = g >> 12, l = g & (LL - 1);
    int lh = l >> 6, lw = l & 63;
    float s = 0.f;
#pragma unroll
    for (int dy = -1; dy <= 1; ++dy)
#pragma unroll
        for (int dx = -1; dx <= 1; ++dx) {
            int h = lh + dy, w = lw + dx;
            if (h >= 0 && h < HH && w >= 0 && w < WW)
                s += E[b * LL + h * WW + w];
        }
    invn[g] = 1.0f / fmaxf(sqrtf(s), 1e-12f);
}

// ---------------------------------------------------------------------------
// Kernel 3: S[l][m] = sum_c Xb[l][c] * Xc[m][c]   (one batch per launch)
// 128x128 tile per block; 8 waves, each wave owns 8 (2 rows x 4 cols) of the
// 64 16x16 subtiles -> 12 ds_load_b128 feed 8 v_wmma per k-step (1.5/wmma).
// ---------------------------------------------------------------------------
__global__ __launch_bounds__(256) void gemm_kernel(
    const u16* __restrict__ Xb, const u16* __restrict__ Xc,
    float* __restrict__ S, int b)
{
    const int tile_m = (blockIdx.x & 31) << 7;   // columns (current loc)
    const int tile_l = (blockIdx.x >> 5) << 7;   // rows (blur loc)
    const int t = threadIdx.x;

    __shared__ u16 As[128 * KPAD];
    __shared__ u16 Bs[128 * KPAD];

    // cooperative load: 4096 uint4 chunks per matrix, 16 per thread
#pragma unroll
    for (int i = 0; i < 16; ++i) {
        int q   = t + 256 * i;
        int row = q >> 5;
        int ko  = (q & 31) * 8;
        *(uint4*)&As[row * KPAD + ko] =
            *(const uint4*)(Xb + ((size_t)(b * LL + tile_l + row)) * CCH + ko);
        *(uint4*)&Bs[row * KPAD + ko] =
            *(const uint4*)(Xc + ((size_t)(b * LL + tile_m + row)) * CCH + ko);
    }
    __syncthreads();

    const int w    = t >> 5;          // wave 0..7
    const int lane = t & 31;
    const int half = lane >> 4;       // 0: lanes 0-15, 1: lanes 16-31
    const int lm   = lane & 15;
    const int rt_base = (w & 3) * 2;  // 2 row subtiles per wave
    const int ct_base = (w >> 2) * 4; // 4 col subtiles per wave

    union Frag { uint4 u[2]; bf16x16 v; };

    floatx8 acc[2][4];
#pragma unroll
    for (int rr = 0; rr < 2; ++rr)
#pragma unroll
        for (int cc = 0; cc < 4; ++cc)
            acc[rr][cc] = (floatx8){0.f, 0.f, 0.f, 0.f, 0.f, 0.f, 0.f, 0.f};

    int arow[2], bcol[4];
#pragma unroll
    for (int rr = 0; rr < 2; ++rr) arow[rr] = (rt_base + rr) * 16 + lm;
#pragma unroll
    for (int cc = 0; cc < 4; ++cc) bcol[cc] = (ct_base + cc) * 16 + lm;

#pragma unroll
    for (int k0 = 0; k0 < 256; k0 += 32) {
        Frag fa[2], fb[4];
        // A (16x32 MxK): lanes 0-15 -> K chunks k0+0..7 / k0+16..23 ; lanes 16-31 +8
#pragma unroll
        for (int rr = 0; rr < 2; ++rr) {
            fa[rr].u[0] = *(const uint4*)&As[arow[rr] * KPAD + k0 + half * 8];
            fa[rr].u[1] = *(const uint4*)&As[arow[rr] * KPAD + k0 + 16 + half * 8];
        }
        // B (32x16 KxN): lanes 0-15 -> K chunks k0+0..7 / k0+16..23 ; lanes 16-31 +8
#pragma unroll
        for (int cc = 0; cc < 4; ++cc) {
            fb[cc].u[0] = *(const uint4*)&Bs[bcol[cc] * KPAD + k0 + half * 8];
            fb[cc].u[1] = *(const uint4*)&Bs[bcol[cc] * KPAD + k0 + 16 + half * 8];
        }
#pragma unroll
        for (int rr = 0; rr < 2; ++rr)
#pragma unroll
            for (int cc = 0; cc < 4; ++cc)
                acc[rr][cc] = __builtin_amdgcn_wmma_f32_16x16x32_bf16(
                    false, fa[rr].v, false, fb[cc].v, (short)0,
                    acc[rr][cc], false, false);
    }

    // D layout: VGPR r -> M = r + 8*half ; N = lane&15
#pragma unroll
    for (int rr = 0; rr < 2; ++rr)
#pragma unroll
        for (int cc = 0; cc < 4; ++cc) {
            int colg = tile_m + (ct_base + cc) * 16 + lm;
#pragma unroll
            for (int r = 0; r < 8; ++r) {
                int rowg = tile_l + (rt_base + rr) * 16 + r + 8 * half;
                S[(size_t)rowg * LL + colg] = acc[rr][cc][r];
            }
        }
}

// ---------------------------------------------------------------------------
// Kernel 4: per current-location m, argmax over l of
//   (sum over valid offsets d of S[l+d, m+d]) * invn[b,l]
// First-occurrence tie-break (matches jnp.argmax). One block per m.
// ---------------------------------------------------------------------------
__global__ __launch_bounds__(256) void argmax_kernel(
    const float* __restrict__ S, const float* __restrict__ invn,
    int* __restrict__ maxidx, int b)
{
    const int m  = blockIdx.x;
    const int mh = m >> 6, mw = m & 63;
    const int t  = threadIdx.x;

    float best = -3.4e38f;
    int   bi   = 0x7FFFFFFF;

    for (int l = t; l < LL; l += 256) {
        int lh = l >> 6, lw = l & 63;
        float T = 0.f;
#pragma unroll
        for (int dy = -1; dy <= 1; ++dy)
#pragma unroll
            for (int dx = -1; dx <= 1; ++dx) {
                int lhh = lh + dy, lww = lw + dx;
                int mhh = mh + dy, mww = mw + dx;
                if (lhh >= 0 && lhh < HH && lww >= 0 && lww < WW &&
                    mhh >= 0 && mhh < HH && mww >= 0 && mww < WW) {
                    T += S[(size_t)(lhh * WW + lww) * LL + (mhh * WW + mww)];
                }
            }
        float val = T * invn[b * LL + l];
        if (val > best) { best = val; bi = l; }   // l ascending per thread
    }

    __shared__ float sv[256];
    __shared__ int   si[256];
    sv[t] = best; si[t] = bi;
    __syncthreads();
    for (int s = 128; s > 0; s >>= 1) {
        if (t < s) {
            float ov = sv[t + s]; int oi = si[t + s];
            if (ov > sv[t] || (ov == sv[t] && oi < si[t])) { sv[t] = ov; si[t] = oi; }
        }
        __syncthreads();
    }
    if (t == 0) maxidx[b * LL + m] = si[0];
}

// ---------------------------------------------------------------------------
// Kernel 5: fold matched sharp-reference patches + residual add.
// out[b,c,y,x] = cur[b,c,y,x] + sum_{i,j in 0..2} refer_pad[c, LH(mi)+i-1, LW(mi)+j-1]
//   where mi = maxidx[b, (y-i+1)*W + (x-j+1)] when that location is valid.
// ---------------------------------------------------------------------------
__global__ __launch_bounds__(256) void fold_kernel(
    const float* __restrict__ cur, const float* __restrict__ refer,
    const int* __restrict__ maxidx, float* __restrict__ out)
{
    int idx = blockIdx.x * 256 + threadIdx.x;   // over BATCH*CCH*LL
    if (idx >= BATCH * CCH * LL) return;
    int b   = idx >> 20;
    int rem = idx & ((1 << 20) - 1);
    int c   = rem >> 12;
    int l   = rem & (LL - 1);
    int y   = l >> 6, x = l & 63;

    float acc = cur[idx];
#pragma unroll
    for (int i = 0; i < 3; ++i)
#pragma unroll
        for (int j = 0; j < 3; ++j) {
            int h = y - i + 1, w = x - j + 1;
            if (h >= 0 && h < HH && w >= 0 && w < WW) {
                int mi = maxidx[b * LL + h * WW + w];
                int rh = (mi >> 6) + i - 1;
                int rw = (mi & 63) + j - 1;
                if (rh >= 0 && rh < HH && rw >= 0 && rw < WW)
                    acc += refer[((size_t)(b * CCH + c)) * LL + rh * WW + rw];
            }
        }
    out[idx] = acc;
}

// ---------------------------------------------------------------------------
extern "C" void kernel_launch(void* const* d_in, const int* in_sizes, int n_in,
                              void* d_out, int out_size, void* d_ws, size_t ws_size,
                              hipStream_t stream)
{
    (void)in_sizes; (void)n_in; (void)out_size; (void)ws_size;
    const float* cur   = (const float*)d_in[0];   // feature_current
    const float* refer = (const float*)d_in[1];   // feature_refer
    const float* blur  = (const float*)d_in[2];   // feature_refer_blur
    float* out = (float*)d_out;

    char* w = (char*)d_ws;
    float* S     = (float*)w;  w += (size_t)LL * LL * 4;          // 64 MB (reused per batch)
    u16*   Xb    = (u16*)w;    w += (size_t)BATCH * LL * CCH * 2; // 8 MB
    u16*   Xc    = (u16*)w;    w += (size_t)BATCH * LL * CCH * 2; // 8 MB
    float* E     = (float*)w;  w += (size_t)BATCH * LL * 4;
    float* invn  = (float*)w;  w += (size_t)BATCH * LL * 4;
    int*   midx  = (int*)w;

    pack_kernel<<<BATCH * (LL / 64), 256, 0, stream>>>(blur, cur, Xb, Xc, E);
    invn_kernel<<<(BATCH * LL + 255) / 256, 256, 0, stream>>>(E, invn);

    for (int b = 0; b < BATCH; ++b) {
        gemm_kernel<<<(LL / 128) * (LL / 128), 256, 0, stream>>>(Xb, Xc, S, b);
        argmax_kernel<<<LL, 256, 0, stream>>>(S, invn, midx, b);
    }

    fold_kernel<<<(BATCH * CCH * LL + 255) / 256, 256, 0, stream>>>(cur, refer, midx, out);
}
